// ODEnet_8787503087642
// MI455X (gfx1250) — compile-verified, matrix operand order checked by reading, not run on hardware
//
#include <hip/hip_runtime.h>
#include <hip/hip_bf16.h>
#include <cstdint>
#include <cstddef>

// ---------------------------------------------------------------------------
// MI455X (gfx1250, wave32) implementation of the learned-flux ODE solver.
// Conv towers run as implicit-GEMM with v_wmma_f32_16x16x32_f16 (f16 data,
// f32 accumulate); activations live in f16 to halve HBM traffic (workload is
// memory-bound: ~1.2 TFLOP vs tens of GB of activation traffic @ 23.3 TB/s).
// Kernels are templated on <CIN,KSIZE> so K-loops fully unroll, A-fragments
// load as 16-byte LDS vectors, and all index math strength-reduces.
// ---------------------------------------------------------------------------

typedef __attribute__((ext_vector_type(16))) _Float16 v16h;
typedef __attribute__((ext_vector_type(8)))  _Float16 v8h;
typedef __attribute__((ext_vector_type(8)))  float    v8f;

union AFrag { v16h v; _Float16 h[16]; v8h q[2]; };
union CFrag { v8f  v; float    f[8];  };

// K element held at fragment slot e for lane-group g (16-bit A/B 16x32 layout)
__device__ __forceinline__ constexpr int kmap_e(int e) { return e < 8 ? e : e + 8; }

// ---------------------------------------------------------------------------
// Implicit-GEMM conv: out[p, co] = act( sum_{ci,dy,dx} in[p+off, ci]*w + b
//                                       (+ residual) ) * scale
// One workgroup per (batch,row). LDS holds KSIZE rows x 130 cols x CIN f16
// (zero-filled halo/tail so the 8th 16-pixel tile reads zeros).
// 8 waves; wave m owns pixels [16m,16m+16); loops N-tiles, unrolled K-chunks.
// ---------------------------------------------------------------------------
template <int CIN, int KSIZE>
__global__ __launch_bounds__(256)
void conv_wmma(const _Float16* __restrict__ in, const _Float16* __restrict__ wb,
               const float* __restrict__ bias, const _Float16* __restrict__ res,
               _Float16* __restrict__ out16, float* __restrict__ out32,
               int H, int W, int CoutPad, int CoutReal, int relu, float scale)
{
    constexpr int HALO = KSIZE >> 1;
    constexpr int KTOT = CIN * KSIZE * KSIZE;
    constexpr int KC   = (KTOT + 31) >> 5;
    constexpr int LOG2CIN = (CIN == 2) ? 1 : (CIN == 16) ? 4 : (CIN == 32) ? 5 : 6;
    constexpr int SELEMS  = KSIZE * 130 * CIN;

    __shared__ _Float16 smem[SELEMS];

    const int tid = threadIdx.x;
    const int b   = blockIdx.x / H;
    const int y   = blockIdx.x % H;

    // zero LDS (halo columns / tail pixels / out-of-range rows read as zero)
    {
        uint32_t* s32 = (uint32_t*)smem;
        #pragma unroll 4
        for (int i = tid; i < SELEMS / 2; i += 256) s32[i] = 0u;
    }
    __syncthreads();

    // stage input rows (channel-last f16)
    #pragma unroll
    for (int r = 0; r < KSIZE; ++r) {
        int yy = y + r - HALO;
        if (yy >= 0 && yy < H) {
            if constexpr (CIN >= 8) {        // 16-byte granules (alignment holds)
                const uint4* src = (const uint4*)(in + (size_t)(b * H + yy) * W * CIN);
                uint4* dst = (uint4*)(smem + (size_t)(r * 130 + HALO) * CIN);
                const int cnt = (W * CIN) >> 3;
                for (int i = tid; i < cnt; i += 256) dst[i] = src[i];
            } else {                          // 4-byte granules (CIN==2)
                const uint32_t* src = (const uint32_t*)(in + (size_t)(b * H + yy) * W * CIN);
                uint32_t* dst = (uint32_t*)(smem + (size_t)(r * 130 + HALO) * CIN);
                const int cnt = (W * CIN) >> 1;
                for (int i = tid; i < cnt; i += 256) dst[i] = src[i];
            }
        }
    }
    __syncthreads();

    const int lane = tid & 31;
    const int wv   = tid >> 5;          // 8 waves -> 8 M tiles (covers 128 >= W=120)
    const int g    = lane >> 4;
    const int nloc = lane & 15;
    const int px0  = wv * 16;
    const int arow = px0 + nloc;        // pixel (A-matrix row) for this lane
    const int NT   = CoutPad >> 4;

    for (int nt = 0; nt < NT; ++nt) {
        CFrag acc;
        #pragma unroll
        for (int r = 0; r < 8; ++r) acc.f[r] = 0.0f;

        const _Float16* wbase = wb + ((size_t)nt * KC * 32 + lane) * 16;

        #pragma unroll
        for (int kc = 0; kc < KC; ++kc) {
            AFrag a;
            if constexpr (CIN >= 8) {
                // each half-group: 8 contiguous K = one filter tap, contiguous ci
                #pragma unroll
                for (int hq = 0; hq < 2; ++hq) {
                    int Kb = (kc << 5) + (hq << 4) + (g << 3);
                    v8h q = {};
                    if (Kb < KTOT) {          // KTOT%8==0 -> group all valid/invalid
                        int pos = Kb >> LOG2CIN;
                        int ci  = Kb & (CIN - 1);
                        int dy  = pos / KSIZE;
                        int dx  = pos - dy * KSIZE;
                        q = *(const v8h*)(smem + (size_t)(dy * 130 + arow + dx) * CIN + ci);
                    }
                    a.q[hq] = q;
                }
            } else {
                #pragma unroll
                for (int e = 0; e < 16; ++e) {
                    int K = (kc << 5) + kmap_e(e) + (g << 3);
                    _Float16 hv = (_Float16)0.0f;
                    if (K < KTOT) {
                        int pos = K >> LOG2CIN;
                        int ci  = K & (CIN - 1);
                        int dy  = pos / KSIZE;
                        int dx  = pos - dy * KSIZE;
                        hv = smem[(size_t)(dy * 130 + arow + dx) * CIN + ci];
                    }
                    a.h[e] = hv;
                }
            }
            // weights pre-packed per lane: one 32B contiguous fragment
            v16h bf = *(const v16h*)(wbase + (size_t)kc * 32 * 16);
            acc.v = __builtin_amdgcn_wmma_f32_16x16x32_f16(
                        false, a.v, false, bf, (short)0, acc.v, false, false);
        }

        // epilogue: D layout -> lane=N col, VGPR r = M row (+8 for hi group)
        int cout = (nt << 4) + nloc;
        if (cout < CoutReal) {
            float bv = bias[cout];
            #pragma unroll
            for (int r = 0; r < 8; ++r) {
                int px = px0 + r + (g << 3);
                if (px < W) {
                    size_t pix = (size_t)(b * H + y) * W + px;
                    float v = acc.f[r] + bv;
                    if (res)  v += (float)res[pix * (size_t)CoutPad + cout];
                    if (relu) v = v > 0.0f ? v : 0.0f;
                    v *= scale;
                    if (out16) out16[pix * (size_t)CoutPad  + cout] = (_Float16)v;
                    if (out32) out32[pix * (size_t)CoutReal + cout] = v;
                }
            }
        }
    }
}

// ---------------------------------------------------------------------------
// Repack f32 OIHW weights into per-lane-contiguous WMMA-B fragments (f16):
// wbuf[nt][kc][lane][e] = w[n = nt*16+lane%16][K = kc*32+kmap(lane/16,e)]
// with K ordered (dy*ksize+dx)*Cin + ci; zero padded.
// ---------------------------------------------------------------------------
__global__ void repack_w(const float* __restrict__ w, _Float16* __restrict__ wbuf,
                         int coutReal, int cin, int k, int NT, int KC)
{
    int idx = blockIdx.x * 256 + threadIdx.x;
    int total = NT * KC * 512;
    if (idx >= total) return;
    int e    = idx & 15;
    int lane = (idx >> 4) & 31;
    int rest = idx >> 9;
    int kc   = rest % KC;
    int nt   = rest / KC;
    int g = lane >> 4;
    int n = (nt << 4) + (lane & 15);
    int K = (kc << 5) + (e < 8 ? e : e + 8) + (g << 3);
    float v = 0.0f;
    int Ktot = cin * k * k;
    if (n < coutReal && K < Ktot) {
        int pos = K / cin, ci = K - pos * cin;
        int dy = pos / k,  dx = pos - dy * k;
        v = w[((size_t)(n * cin + ci) * k + dy) * k + dx];
    }
    wbuf[idx] = (_Float16)v;
}

// ---------------------------------------------------------------------------
// Build the four 2-channel direction inputs (f16, channel-last, edge-clamped)
// ---------------------------------------------------------------------------
__global__ void build_dirs(const float* __restrict__ z,
                           _Float16* dl, _Float16* dr, _Float16* dt, _Float16* db,
                           int H, int W, int total)
{
    int idx = blockIdx.x * 256 + threadIdx.x;
    if (idx >= total) return;
    int x = idx % W; int t = idx / W; int y = t % H; int b = t / H;
    const float* zp = z + (size_t)b * H * W;
    float um = zp[y * W + x];
    float ul = zp[y * W + (x > 0     ? x - 1 : 0)];
    float ur = zp[y * W + (x < W - 1 ? x + 1 : W - 1)];
    float ut = zp[(y > 0     ? y - 1 : 0)     * W + x];
    float ub = zp[(y < H - 1 ? y + 1 : H - 1) * W + x];
    dl[2*idx] = (_Float16)ul; dl[2*idx+1] = (_Float16)um;   // concat(ul, um)
    dr[2*idx] = (_Float16)um; dr[2*idx+1] = (_Float16)ur;   // concat(um, ur)
    dt[2*idx] = (_Float16)ut; dt[2*idx+1] = (_Float16)um;   // concat(ut, um)
    db[2*idx] = (_Float16)um; db[2*idx+1] = (_Float16)ub;   // concat(um, ub)
}

// ---------------------------------------------------------------------------
// CONP==1 flux: edge contribution = du * sum(L^2 * (lam +/- |lam|)) / sum(L^2)
// then z' = z + h * (-(sum)/(2*DX))
// ---------------------------------------------------------------------------
__device__ __forceinline__ float edge_term(const float* __restrict__ L,
                                           const float* __restrict__ lam,
                                           size_t pix, float du, float sgn)
{
    float num = 0.0f, den = 0.0f;
    #pragma unroll
    for (int k = 0; k < 8; ++k) {
        float Lk = L[pix * 8 + k];
        float lk = lam[pix * 8 + k];
        float lt = lk + sgn * fabsf(lk);
        num += Lk * Lk * lt;
        den += Lk * Lk;
    }
    return du * num / den;
}

__global__ void flux_euler(const float* __restrict__ z,
                           const float* laml, const float* lamr,
                           const float* lamt, const float* lamb,
                           const float* Ll_,  const float* Lr_,
                           const float* Lt_,  const float* Lb_,
                           float* __restrict__ zout,
                           int H, int W, int total, float hstep)
{
    int idx = blockIdx.x * 256 + threadIdx.x;
    if (idx >= total) return;
    int x = idx % W; int t = idx / W; int y = t % H; int b = t / H;
    const float* zp = z + (size_t)b * H * W;
    float um = zp[y * W + x];
    float ul = zp[y * W + (x > 0     ? x - 1 : 0)];
    float ur = zp[y * W + (x < W - 1 ? x + 1 : W - 1)];
    float ut = zp[(y > 0     ? y - 1 : 0)     * W + x];
    float ub = zp[(y < H - 1 ? y + 1 : H - 1) * W + x];
    size_t p = (size_t)idx;
    float s = edge_term(Lr_, lamr, p, ur - um, -1.0f)
            + edge_term(Ll_, laml, p, um - ul, +1.0f)
            + edge_term(Lt_, lamt, p, um - ut, +1.0f)
            + edge_term(Lb_, lamb, p, ub - um, -1.0f);
    zout[idx] = um + hstep * (-s / (2.0f * 0.01f));
}

// ---------------------------------------------------------------------------
// Ghost-cell wrap: out[r] = in[f(r)], f(r) = r<g ? r+H-2g : (r<H-g ? r : r-H+2g)
// ---------------------------------------------------------------------------
__global__ void boundary(const float* __restrict__ zin, float* __restrict__ zout,
                         int H, int W, int gst, int total)
{
    int idx = blockIdx.x * 256 + threadIdx.x;
    if (idx >= total) return;
    int x = idx % W; int t = idx / W; int y = t % H; int b = t / H;
    int sy = (y < gst) ? y + (H - 2 * gst) : (y < H - gst ? y : y - (H - 2 * gst));
    int sx = (x < gst) ? x + (W - 2 * gst) : (x < W - gst ? x : x - (W - 2 * gst));
    zout[idx] = zin[(size_t)(b * H + sy) * W + sx];
}

// ===========================================================================
// Host orchestration
// ===========================================================================
namespace {
struct Conv { const float* w; const float* b; int cin, cout, k; _Float16* wb; };
struct Blk  { Conv c1, c2, cs; bool hasWs; };
struct Tow  { Blk blk[6]; Conv fin; };
}

extern "C" void kernel_launch(void* const* d_in, const int* in_sizes, int n_in,
                              void* d_out, int out_size, void* d_ws, size_t ws_size,
                              hipStream_t stream)
{
    (void)in_sizes; (void)n_in; (void)out_size; (void)ws_size;

    const int B = 32, H = 120, W = 120, GST = 10;
    const int P = B * H * W;
    const float hstep = 0.5f;   // t1_t0=1, EPS=0.5 -> n_steps=2, h=0.5

    // ---- parse params; inner dicts flattened in jax-sorted key order ----
    static const int s_cin[6]  = { 2, 16, 16, 32, 64, 64 };
    static const int s_cout[6] = {16, 16, 32, 64, 64, 64 };
    static const int s_k[6]    = { 3,  3,  3,  3,  3,  1 };

    int ip = 2;                 // d_in[0]=z0, d_in[1]=t1_t0
    auto nextp = [&]() -> const float* { return (const float*)d_in[ip++]; };

    Tow tow[4];                 // 0: lam_v, 1: lam_h, 2: L_v, 3: L_h
    for (int t = 0; t < 4; ++t) {
        for (int i = 0; i < 6; ++i) {
            bool ws = (s_cin[i] != s_cout[i]);
            const float *b1, *b2, *bs = nullptr, *w1, *w2, *wp = nullptr;
            b1 = nextp(); b2 = nextp();
            if (ws) bs = nextp();
            w1 = nextp(); w2 = nextp();
            if (ws) wp = nextp();
            Blk& bk = tow[t].blk[i];
            bk.c1 = { w1, b1, s_cin[i],  s_cout[i], s_k[i], nullptr };
            bk.c2 = { w2, b2, s_cout[i], s_cout[i], 1,      nullptr };
            bk.cs = { wp, bs, s_cin[i],  s_cout[i], 1,      nullptr };
            bk.hasWs = ws;
        }
        const float* fb = nextp();
        const float* fw = nextp();
        tow[t].fin = { fw, fb, 64, 8, 1, nullptr };
    }

    // ---- carve workspace ----
    char* ws = (char*)d_ws;
    size_t off = 0;
    auto carve = [&](size_t bytes) -> void* {
        off = (off + 255) & ~(size_t)255;
        void* p = ws + off;
        off += bytes;
        return p;
    };

    _Float16* act0 = (_Float16*)carve((size_t)P * 64 * 2);
    _Float16* act1 = (_Float16*)carve((size_t)P * 64 * 2);
    _Float16* act2 = (_Float16*)carve((size_t)P * 64 * 2);
    _Float16* dirs[4];
    for (int i = 0; i < 4; ++i) dirs[i] = (_Float16*)carve((size_t)P * 2 * 2);
    float *lamB[4], *LB[4];
    for (int i = 0; i < 4; ++i) lamB[i] = (float*)carve((size_t)P * 8 * 4);
    for (int i = 0; i < 4; ++i) LB[i]   = (float*)carve((size_t)P * 8 * 4);
    float* ztmp = (float*)carve((size_t)P * 4);
    float* zA   = (float*)carve((size_t)P * 4);

    // ---- repack all weights to f16 WMMA-B fragments (once per launch) ----
    auto repack = [&](Conv& c) {
        if (!c.w) return;
        int cp = (c.cout + 15) & ~15;
        int NT = cp >> 4;
        int KC = (c.cin * c.k * c.k + 31) >> 5;
        c.wb = (_Float16*)carve((size_t)NT * KC * 512 * 2);
        int total = NT * KC * 512;
        repack_w<<<(total + 255) / 256, 256, 0, stream>>>(c.w, c.wb, c.cout, c.cin, c.k, NT, KC);
    };
    for (int t = 0; t < 4; ++t) {
        for (int i = 0; i < 6; ++i) {
            repack(tow[t].blk[i].c1);
            repack(tow[t].blk[i].c2);
            if (tow[t].blk[i].hasWs) repack(tow[t].blk[i].cs);
        }
        repack(tow[t].fin);
    }

    // ---- conv dispatch over <CIN,KSIZE> instantiations ----
    auto conv = [&](const _Float16* in, Conv& c, const _Float16* res,
                    _Float16* o16, float* o32, int relu, float scale) {
        int cp = (c.cout + 15) & ~15;
        dim3 grid(B * H);
        #define LAUNCH_CONV(CI, KS)                                             \
            conv_wmma<CI, KS><<<grid, 256, 0, stream>>>(in, c.wb, c.b, res,     \
                o16, o32, H, W, cp, c.cout, relu, scale)
        if (c.k == 3) {
            if      (c.cin == 2)  LAUNCH_CONV(2, 3);
            else if (c.cin == 16) LAUNCH_CONV(16, 3);
            else if (c.cin == 32) LAUNCH_CONV(32, 3);
            else                  LAUNCH_CONV(64, 3);
        } else {
            if      (c.cin == 2)  LAUNCH_CONV(2, 1);
            else if (c.cin == 16) LAUNCH_CONV(16, 1);
            else if (c.cin == 32) LAUNCH_CONV(32, 1);
            else                  LAUNCH_CONV(64, 1);
        }
        #undef LAUNCH_CONV
    };

    auto run_tower = [&](const _Float16* dir, Tow& T, float* out, float scale) {
        const _Float16* cur = dir;
        for (int i = 0; i < 6; ++i) {
            Blk& bk = T.blk[i];
            conv(cur, bk.c1, nullptr, act0, nullptr, 1, 1.0f);          // relu(conv kxk)
            const _Float16* res;
            if (bk.hasWs) { conv(cur, bk.cs, nullptr, act1, nullptr, 0, 1.0f); res = act1; }
            else          { res = cur; }                                 // identity shortcut
            conv(act0, bk.c2, res, act2, nullptr, 1, 1.0f);              // relu(conv1x1 + sc)
            cur = act2;   // elementwise in-place residual is same-thread read-before-write
        }
        conv(cur, T.fin, nullptr, nullptr, out, 0, scale);               // 64->8 fp32 out
    };

    auto step = [&](const float* zin, float* zout) {
        build_dirs<<<(P + 255) / 256, 256, 0, stream>>>(zin, dirs[0], dirs[1], dirs[2], dirs[3], H, W, P);
        run_tower(dirs[0], tow[1], lamB[0], 0.1f);   // left   : lam_h
        run_tower(dirs[0], tow[3], LB[0],   1.0f);   // left   : L_h
        run_tower(dirs[1], tow[1], lamB[1], 0.1f);   // right  : lam_h
        run_tower(dirs[1], tow[3], LB[1],   1.0f);   // right  : L_h
        run_tower(dirs[2], tow[0], lamB[2], 0.1f);   // top    : lam_v
        run_tower(dirs[2], tow[2], LB[2],   1.0f);   // top    : L_v
        run_tower(dirs[3], tow[0], lamB[3], 0.1f);   // bottom : lam_v
        run_tower(dirs[3], tow[2], LB[3],   1.0f);   // bottom : L_v
        flux_euler<<<(P + 255) / 256, 256, 0, stream>>>(zin,
            lamB[0], lamB[1], lamB[2], lamB[3], LB[0], LB[1], LB[2], LB[3],
            ztmp, H, W, P, hstep);
        boundary<<<(P + 255) / 256, 256, 0, stream>>>(ztmp, zout, H, W, GST, P);
    };

    step((const float*)d_in[0], zA);
    step(zA, (float*)d_out);
}
